// SparseResUNet_76673756168766
// MI455X (gfx1250) — matrix operand order repaired and checked by reading, not compiled, not run on hardware
//
#include <hip/hip_runtime.h>

// ---------------------------------------------------------------------------
// Sparse ResUNet on MI455X (gfx1250).
// Gather-GEMM-scatter sparse conv with v_wmma_f32_16x16x32_bf16.
// Block = NW wave32 waves (NW=2 for Cout<=32 layers, else 4); a block owns
// 16 points x (16*NW) out-channels.  Per kernel-offset k the 16 gathered bf16
// rows are staged in double-buffered LDS, software-pipelined:
//     wait(async) ; barrier ; issue fill(k+1) ; compute(k)
// so the async global->LDS copy of the next offset overlaps this offset's
// WMMA loop.  Fill uses GLOBAL_LOAD_ASYNC_TO_LDS_B64 when the toolchain
// exposes the builtin (ASYNCcnt-tracked, no VGPR round-trip), else a
// vectorized load + ds_store fallback.
//   A: two ds_load_b128 per lane from LDS
//   B: two 16B contiguous loads per lane from bf16 fragment-major prepacked
//      weights [k][chunk][half][nCol][16]  (prepacked once, on-stream)
// Fragment packing: contiguous k = 16*h + s in both A and B.
// ---------------------------------------------------------------------------

typedef __attribute__((ext_vector_type(16))) __bf16 v16bf;
typedef __attribute__((ext_vector_type(8)))  __bf16 v8bf;
typedef __attribute__((ext_vector_type(8)))  float  v8f;
typedef __attribute__((vector_size(2 * sizeof(int)))) int v2i;

#define MAXCP 384   // largest (padded) Cin in the network

#if __has_builtin(__builtin_amdgcn_global_load_async_to_lds_b64)
#define ASYNC_OK 1
#else
#define ASYNC_OK 0
#endif

#if ASYNC_OK
__device__ __forceinline__ void wait_async0() {
#if __has_builtin(__builtin_amdgcn_s_wait_asynccnt)
  __builtin_amdgcn_s_wait_asynccnt(0);
#else
  asm volatile("s_wait_asynccnt 0x0" ::: "memory");
#endif
}
#endif

__device__ __forceinline__ __bf16 f2bf(float f) {
  union { float f; unsigned u; } x; x.f = f;
  unsigned r = x.u + 0x7fffu + ((x.u >> 16) & 1u);   // round-to-nearest-even
  unsigned short h = (unsigned short)(r >> 16);
  __bf16 o;
  __builtin_memcpy(&o, &h, sizeof(o));
  return o;
}

// y[nOut,Cout] = sum_k f[nbr[n,k],:] @ w[k]   (nbr<0 => zero row)
// f: bf16, row stride Cp (multiple of 32, zero padded)
// wp: prepacked bf16 weights, [k][q][h][n][i] with i contiguous (16)
// nbr == nullptr => identity gather (K==1): 1x1 shortcut conv.
template <int NW>
__global__ __launch_bounds__(NW * 32)
void sconv_wmma(const __bf16* __restrict__ f, const int* __restrict__ nbr,
                const __bf16* __restrict__ wp, float* __restrict__ y,
                int nOut, int Cp, int Cout, int K) {
  constexpr int LCP  = (NW == 2) ? 32 : MAXCP;  // LDS row capacity
  constexpr int TPR  = 2 * NW;                  // fill threads per row
  constexpr int SPAN = 16 / NW;                 // bf16 elems per thread/chunk
  __shared__ __align__(16) __bf16 Asm[2 * 16 * LCP];

  const int t     = threadIdx.x;
  const int lane  = t & 31;
  const int wv    = t >> 5;                     // wave id within block
  const int mRow  = lane & 15;
  const int h     = lane >> 4;                  // half-wave selector
  const int mBase = blockIdx.x * 16;
  const int nBase = (blockIdx.y * NW + wv) * 16;
  const bool nOk  = (nBase < Cout);             // Cout is a multiple of 16
  const int nCol  = nBase + mRow;
  const int QC    = Cp >> 5;                    // 32-channel chunks

  const int fr  = t / TPR;                      // fill row 0..15
  const int fcb = (t % TPR) * SPAN;             // fill col base within chunk
  const int gm  = mBase + fr;

  auto fill = [&](int kk, __bf16* At) {
    int grow = -1;
    if (gm < nOut) grow = nbr ? nbr[(long long)gm * K + kk] : gm;
    if (nbr && kk + 1 < K && gm < nOut)
      __builtin_prefetch(&nbr[(long long)gm * K + kk + 1], 0, 0);  // global_prefetch_b8
    const __bf16* frow = f + (long long)grow * Cp;
    for (int q = 0; q < QC; ++q) {
#pragma unroll
      for (int e = 0; e < SPAN; e += 4) {
        int c = q * 32 + fcb + e;
#if ASYNC_OK
        if (grow >= 0) {
          __builtin_amdgcn_global_load_async_to_lds_b64(
              (v2i*)(frow + c), (v2i*)(At + fr * Cp + c), 0, 0);
        } else {
          uint2 z; z.x = 0u; z.y = 0u;
          *(uint2*)(At + fr * Cp + c) = z;
        }
#else
        uint2 v; v.x = 0u; v.y = 0u;
        if (grow >= 0) v = *(const uint2*)(frow + c);
        *(uint2*)(At + fr * Cp + c) = v;
#endif
      }
    }
  };

  v8f acc = {};
  fill(0, &Asm[0]);
  for (int k = 0; k < K; ++k) {
    __bf16* At = &Asm[(k & 1) * (16 * LCP)];
#if ASYNC_OK
    wait_async0();
#endif
    __syncthreads();                       // buffer k complete for all threads
    if (k + 1 < K) fill(k + 1, &Asm[((k + 1) & 1) * (16 * LCP)]);  // overlap

    if (nOk) {                             // wave-uniform: EXEC stays all-ones
      const __bf16* arow = At + mRow * Cp;
      const __bf16* wk   = wp + ((long long)k * QC * 2 * Cout) * 16;
      for (int q = 0; q < QC; ++q) {
        // A fragment: two ds_load_b128
        v8bf lo = *(const v8bf*)(arow + q * 32 + 16 * h);
        v8bf hi = *(const v8bf*)(arow + q * 32 + 16 * h + 8);
        v16bf a = __builtin_shufflevector(lo, hi, 0, 1, 2, 3, 4, 5, 6, 7,
                                          8, 9, 10, 11, 12, 13, 14, 15);
        // B fragment: two contiguous 16B loads from prepacked weights
        const __bf16* wb = wk + (((long long)q * 2 + h) * Cout + nCol) * 16;
        v8bf b0 = *(const v8bf*)wb;
        v8bf b1 = *(const v8bf*)(wb + 8);
        v16bf b = __builtin_shufflevector(b0, b1, 0, 1, 2, 3, 4, 5, 6, 7,
                                          8, 9, 10, 11, 12, 13, 14, 15);
        acc = __builtin_amdgcn_wmma_f32_16x16x32_bf16(
            false, a, false, b, (short)0, acc, false, false);
      }
    }
  }
  // ---- store D ----
  if (nOk) {
#pragma unroll
    for (int r = 0; r < 8; ++r) {
      int mm = mBase + r + h * 8;
      if (mm < nOut) y[(long long)mm * Cout + nCol] = acc[r];
    }
  }
}

// prepack f32 weights [K,Cin,Cout] -> bf16 fragment-major [k][q][h][n][16]
__global__ void prepack_w(const float* __restrict__ w, __bf16* __restrict__ wp,
                          int K, int Cin, int Cout, int Cp) {
  int QC = Cp >> 5;
  long long total = (long long)K * QC * 2 * Cout * 16;
  long long i = (long long)blockIdx.x * blockDim.x + threadIdx.x;
  if (i >= total) return;
  int ii = (int)(i & 15);
  long long r = i >> 4;
  int n = (int)(r % Cout); r /= Cout;
  int hh = (int)(r & 1);   r >>= 1;
  int q = (int)(r % QC);
  int k = (int)(r / QC);
  int c = q * 32 + hh * 16 + ii;
  float v = (c < Cin) ? w[((long long)k * Cin + c) * Cout + n] : 0.0f;
  wp[i] = f2bf(v);
}

// f32 [n,C] -> bf16 [n,Cp], zero padded
__global__ void cvt_pad(const float* __restrict__ x, __bf16* __restrict__ o,
                        int n, int C, int Cp) {
  long long i = (long long)blockIdx.x * blockDim.x + threadIdx.x;
  if (i >= (long long)n * Cp) return;
  int c = (int)(i % Cp);
  long long r = i / Cp;
  o[i] = f2bf(c < C ? x[r * C + c] : 0.0f);
}

__global__ void zero_f(float* p, int n) {
  int i = blockIdx.x * blockDim.x + threadIdx.x;
  if (i < n) p[i] = 0.0f;
}

// per-channel sum / sum-of-squares into stats[0..C-1] / stats[C..2C-1]
__global__ void col_stats(const float* __restrict__ y, float* __restrict__ stats,
                          int n, int C) {
  __shared__ float ssum[512];
  __shared__ float ssq[512];
  for (int i = threadIdx.x; i < C; i += blockDim.x) { ssum[i] = 0.f; ssq[i] = 0.f; }
  __syncthreads();
  int r0 = blockIdx.x * 64;
  int r1 = r0 + 64; if (r1 > n) r1 = n;
  int total = (r1 - r0) * C;
  for (int idx = threadIdx.x; idx < total; idx += blockDim.x) {
    int r = r0 + idx / C, c = idx % C;
    float v = y[(long long)r * C + c];
    atomicAdd(&ssum[c], v);
    atomicAdd(&ssq[c], v * v);
  }
  __syncthreads();
  for (int i = threadIdx.x; i < C; i += blockDim.x) {
    atomicAdd(&stats[i], ssum[i]);
    atomicAdd(&stats[C + i], ssq[i]);
  }
}

// out = (relu?)( g*(y-m)*rsqrt(v+eps)+b + (s? s : 0) ); optional bf16 mirror
__global__ void bn_apply(const float* __restrict__ y, const float* __restrict__ stats,
                         const float* __restrict__ g, const float* __restrict__ b,
                         const float* __restrict__ s, float* __restrict__ out,
                         __bf16* __restrict__ obf, int n, int C, int relu) {
  long long i = (long long)blockIdx.x * blockDim.x + threadIdx.x;
  if (i >= (long long)n * C) return;
  int c = (int)(i % C);
  float inv_n = 1.0f / (float)n;
  float m  = stats[c] * inv_n;
  float vv = stats[C + c] * inv_n - m * m;
  float x = g[c] * (y[i] - m) * rsqrtf(vv + 1e-5f) + b[c];
  if (s) x += s[i];
  x = relu ? fmaxf(x, 0.0f) : x;
  out[i] = x;
  if (obf) obf[i] = f2bf(x);
}

// concat -> bf16 (concat output only ever feeds convs)
__global__ void concat2bf(const float* __restrict__ a, const float* __restrict__ b,
                          __bf16* __restrict__ out, int n, int ca, int cb) {
  long long i = (long long)blockIdx.x * blockDim.x + threadIdx.x;
  int cc = ca + cb;
  if (i >= (long long)n * cc) return;
  int r = (int)(i / cc), c = (int)(i % cc);
  float v = (c < ca) ? a[(long long)r * ca + c] : b[(long long)r * cb + (c - ca)];
  out[i] = f2bf(v);
}

// ---------------------------------------------------------------------------
struct Conv { const float *w, *g, *b; int K, Cin, Cout; __bf16* wp; int Cp; };

extern "C" void kernel_launch(void* const* d_in, const int* in_sizes, int n_in,
                              void* d_out, int out_size, void* d_ws, size_t ws_size,
                              hipStream_t stream) {
  (void)out_size; (void)ws_size;
  // ---- parse flattened params (insertion order of _make_params) ----
  int p = 1;
  auto take = [&](int K, int ci, int co) {
    Conv c{(const float*)d_in[p], (const float*)d_in[p + 1], (const float*)d_in[p + 2],
           K, ci, co, nullptr, (ci + 31) & ~31};
    p += 3;
    return c;
  };
  const int CSa[9] = {32, 32, 64, 128, 256, 256, 128, 96, 96};
  Conv stem0 = take(27, 4, 32);
  Conv stem1 = take(27, 32, 32);
  Conv sd[5], sr1a[5], sr1b[5], sr1s[5], sr2a[5], sr2b[5]; bool hasS[5] = {};
  for (int i = 1; i <= 4; ++i) {
    int ci = CSa[i - 1], co = CSa[i];
    sd[i]   = take(8, ci, ci);
    sr1a[i] = take(27, ci, co);
    sr1b[i] = take(27, co, co);
    hasS[i] = (ci != co);
    if (hasS[i]) sr1s[i] = take(1, ci, co);
    sr2a[i] = take(27, co, co);
    sr2b[i] = take(27, co, co);
  }
  const int ups[4][3] = {{256, 256, 128}, {256, 128, 64}, {128, 96, 32}, {96, 96, 32}};
  Conv ut[5], ur1a[5], ur1b[5], ur1s[5], ur2a[5], ur2b[5]; int usk[5] = {};
  for (int j = 1; j <= 4; ++j) {
    int ci = ups[j - 1][0], co = ups[j - 1][1], sk = ups[j - 1][2];
    ut[j]   = take(8, ci, co);
    ur1a[j] = take(27, co + sk, co);
    ur1b[j] = take(27, co, co);
    ur1s[j] = take(1, co + sk, co);
    ur2a[j] = take(27, co, co);
    ur2b[j] = take(27, co, co);
    usk[j] = sk;
  }
  // ---- rulebooks: last 13 inputs ----
  const int nb = n_in - 13;
  const int* nbr3[5]; const int* nbrD[5] = {}; const int* nbrU[5] = {};
  int N[5];
  for (int l = 0; l < 5; ++l) { nbr3[l] = (const int*)d_in[nb + l]; N[l] = in_sizes[nb + l] / 27; }
  for (int l = 1; l <= 4; ++l) nbrD[l] = (const int*)d_in[nb + 4 + l];
  for (int l = 1; l <= 4; ++l) nbrU[l] = (const int*)d_in[nb + 8 + l];
  const float* feats = (const float*)d_in[0];
  const int N0 = N[0];

  // ---- workspace bump allocator ----
  char* wsp = (char*)d_ws;
  auto allocB = [&](long long bytes) {
    void* r = wsp; wsp += ((bytes + 255) & ~255LL); return r;
  };
  auto allocF = [&](long long n) { return (float*)allocB(n * 4); };
  auto allocH = [&](long long n) { return (__bf16*)allocB(n * 2); };

  float* skip[5]; __bf16* skipbf[5];
  for (int l = 0; l < 5; ++l) {
    skip[l]   = allocF((long long)N[l] * CSa[l]);
    skipbf[l] = allocH((long long)N[l] * CSa[l]);
  }
  float*  A     = allocF((long long)N0 * 128);
  float*  B     = allocF((long long)N0 * 128);
  float*  T1    = allocF((long long)N0 * 96);
  float*  T2    = allocF((long long)N0 * 96);
  float*  S     = allocF((long long)N0 * 96);
  __bf16* Abf   = allocH((long long)N0 * 128);
  __bf16* Bbf   = allocH((long long)N0 * 128);
  __bf16* T1bf  = allocH((long long)N0 * 96);
  __bf16* CATbf = allocH((long long)N0 * 128);
  __bf16* FB    = allocH((long long)N0 * 32);
  float*  stats = allocF(1024);

  // ---- prepack all weights to bf16 fragment-major (on-stream, cheap) ----
  Conv* all[64]; int na = 0;
  all[na++] = &stem0; all[na++] = &stem1;
  for (int i = 1; i <= 4; ++i) {
    all[na++] = &sd[i]; all[na++] = &sr1a[i]; all[na++] = &sr1b[i];
    if (hasS[i]) all[na++] = &sr1s[i];
    all[na++] = &sr2a[i]; all[na++] = &sr2b[i];
  }
  for (int j = 1; j <= 4; ++j) {
    all[na++] = &ut[j]; all[na++] = &ur1a[j]; all[na++] = &ur1b[j];
    all[na++] = &ur1s[j]; all[na++] = &ur2a[j]; all[na++] = &ur2b[j];
  }
  for (int q = 0; q < na; ++q) {
    Conv& c = *all[q];
    long long tot = (long long)c.K * (c.Cp >> 5) * 2 * c.Cout * 16;
    c.wp = allocH(tot);
    prepack_w<<<dim3((unsigned)((tot + 255) / 256)), dim3(256), 0, stream>>>(
        c.w, c.wp, c.K, c.Cin, c.Cout, c.Cp);
  }

  // ---- launch helpers ----
  auto sconv = [&](const __bf16* fbf, const int* nbr, const Conv& c, float* y, int nOut) {
    if (c.Cout <= 32 && c.Cp <= 32) {
      dim3 grid((nOut + 15) / 16, (c.Cout + 31) / 32);
      sconv_wmma<2><<<grid, dim3(64), 0, stream>>>(fbf, nbr, c.wp, y, nOut, c.Cp, c.Cout, c.K);
    } else {
      dim3 grid((nOut + 15) / 16, (c.Cout + 63) / 64);
      sconv_wmma<4><<<grid, dim3(128), 0, stream>>>(fbf, nbr, c.wp, y, nOut, c.Cp, c.Cout, c.K);
    }
  };
  auto bn = [&](const float* y, const Conv& c, const float* s,
                float* out, __bf16* obf, int n, int relu) {
    zero_f<<<dim3((2 * c.Cout + 255) / 256), dim3(256), 0, stream>>>(stats, 2 * c.Cout);
    col_stats<<<dim3((n + 63) / 64), dim3(256), 0, stream>>>(y, stats, n, c.Cout);
    long long tot = (long long)n * c.Cout;
    bn_apply<<<dim3((unsigned)((tot + 255) / 256)), dim3(256), 0, stream>>>(
        y, stats, c.g, c.b, s, out, obf, n, c.Cout, relu);
  };
  auto cbr = [&](const __bf16* fbf, const int* nbr, const Conv& c,
                 float* dst, __bf16* dstbf, int n) {
    sconv(fbf, nbr, c, T1, n);
    bn(T1, c, nullptr, dst, dstbf, n, 1);
  };
  auto res = [&](const float* f32p, const __bf16* fbf, const int* nbr,
                 const Conv& ca, const Conv& cb, const Conv* cs,
                 float* dst, __bf16* dstbf, int n) {
    sconv(fbf, nbr, ca, T1, n);
    bn(T1, ca, nullptr, T1, T1bf, n, 1);           // relu(bn(conv_a)) in-place
    sconv(T1bf, nbr, cb, T2, n);
    const float* sp = f32p;
    if (cs) { sconv(fbf, nullptr, *cs, S, n); bn(S, *cs, nullptr, S, nullptr, n, 0); sp = S; }
    bn(T2, cb, sp, dst, dstbf, n, 1);              // relu(bn(conv_b) + shortcut)
  };

  // ---- forward ----
  {
    long long tot = (long long)N0 * 32;
    cvt_pad<<<dim3((unsigned)((tot + 255) / 256)), dim3(256), 0, stream>>>(
        feats, FB, N0, 4, 32);
  }
  cbr(FB,  nbr3[0], stem0, A, Abf, N0);
  cbr(Abf, nbr3[0], stem1, skip[0], skipbf[0], N0);
  const float* x = skip[0]; const __bf16* xbf = skipbf[0];
  for (int i = 1; i <= 4; ++i) {
    cbr(xbf, nbrD[i], sd[i], A, Abf, N[i]);
    res(A, Abf, nbr3[i], sr1a[i], sr1b[i], hasS[i] ? &sr1s[i] : nullptr, B, Bbf, N[i]);
    res(B, Bbf, nbr3[i], sr2a[i], sr2b[i], nullptr, skip[i], skipbf[i], N[i]);
    x = skip[i]; xbf = skipbf[i];
  }
  for (int j = 1; j <= 4; ++j) {
    int l = 4 - j;                                  // fine level of this stage
    cbr(xbf, nbrU[l + 1], ut[j], A, nullptr, N[l]); // deconv -> A (f32 only)
    int co = ut[j].Cout, sk = usk[j], cc = co + sk;
    long long tot = (long long)N[l] * cc;
    concat2bf<<<dim3((unsigned)((tot + 255) / 256)), dim3(256), 0, stream>>>(
        A, skip[l], CATbf, N[l], co, sk);
    // res1: shortcut conv always present, f32 input unused
    res(nullptr, CATbf, nbr3[l], ur1a[j], ur1b[j], &ur1s[j], A, Abf, N[l]);
    float* dst = (j == 4) ? (float*)d_out : B;
    __bf16* dbf = (j == 4) ? nullptr : Bbf;
    res(A, Abf, nbr3[l], ur2a[j], ur2b[j], nullptr, dst, dbf, N[l]);
    x = dst; xbf = Bbf;
  }
}
